// LearnableUpsamplingLayer_343597384307
// MI455X (gfx1250) — compile-verified, matrix operand order checked by reading, not run on hardware
//
#include <hip/hip_runtime.h>
#include <hip/hip_bf16.h>

typedef __attribute__((ext_vector_type(16))) _Float16 v16h;
typedef __attribute__((ext_vector_type(8)))  _Float16 v8h;
typedef __attribute__((ext_vector_type(8)))  float    v8f;
typedef int v4i __attribute__((vector_size(16)));

#define T_TEXT 1024
#define C_DIM  192

#ifndef __has_builtin
#define __has_builtin(x) 0
#endif
#if __has_builtin(__builtin_amdgcn_global_load_async_to_lds_b128) && \
    __has_builtin(__builtin_amdgcn_s_wait_asynccnt)
#define USE_ASYNC_LDS 1
#else
#define USE_ASYNC_LDS 0
#endif

__device__ __forceinline__ float silu_f(float v){ return v / (1.f + __expf(-v)); }

// Load one WMMA 16-bit operand in "NT" form: matrix stored row-major [rows][ldk],
// rows are the M (or N) index, K contiguous. Per CDNA5 ISA 16-bit A layout:
// lanes 0-15 hold K = {k0..k0+7, k0+16..k0+23}, lanes 16-31 hold K = {k0+8..k0+15, k0+24..k0+31}.
__device__ __forceinline__ v16h load_nt16(const _Float16* base, int row, int ldk, int k0, int lane){
  const _Float16* p = base + (size_t)row * ldk + k0 + ((lane >> 4) << 3);
  v8h lo = *(const v8h*)(p);
  v8h hi = *(const v8h*)(p + 16);
  return __builtin_shufflevector(lo, hi, 0,1,2,3,4,5,6,7,8,9,10,11,12,13,14,15);
}

__device__ __forceinline__ v8f wmma_f16(v16h a, v16h b, v8f c){
  return __builtin_amdgcn_wmma_f32_16x16x32_f16(false, a, false, b, (short)0, c, false, false);
}

#if USE_ASYNC_LDS
__device__ __forceinline__ void async_g2l_b128(const void* gsrc, void* ldst){
  __builtin_amdgcn_global_load_async_to_lds_b128(
      (__attribute__((address_space(1))) v4i*)gsrc,
      (__attribute__((address_space(3))) v4i*)ldst, 0, 0);
}
#endif

// ---------------- duration prefix scan: Ed = cumsum(d), Sd = Ed - d ----------------
__global__ void k_scan(const int* __restrict__ d, float* __restrict__ Sd, float* __restrict__ Ed){
  __shared__ float s[T_TEXT];
  const int t = threadIdx.x;
  const float v = (float)d[t];
  s[t] = v;
  __syncthreads();
  for (int off = 1; off < T_TEXT; off <<= 1){
    float add = (t >= off) ? s[t - off] : 0.f;
    __syncthreads();
    s[t] += add;
    __syncthreads();
  }
  Ed[t] = s[t];
  Sd[t] = s[t] - v;
}

// ---------------- f32 -> f16 convert / transpose ----------------
__global__ void k_convert_f16(const float* __restrict__ src, _Float16* __restrict__ dst, int n){
  int i = blockIdx.x * blockDim.x + threadIdx.x;
  if (i < n) dst[i] = (_Float16)src[i];
}
// src row-major [R][Cc] -> dst row-major [Cc][R]
__global__ void k_transpose_f16(const float* __restrict__ src, _Float16* __restrict__ dst, int R, int Cc){
  int i = blockIdx.x * blockDim.x + threadIdx.x;
  if (i < R * Cc){
    int r = i / Cc, c = i % Cc;
    dst[(size_t)c * R + r] = (_Float16)src[i];
  }
}

// ---------------- proj_in: x[t][n] = silu( sum_c H[c][t]*W_in[c][n] + b_in[n] ) ----------------
// A = Ht16 [1024][192] f16, Bt = WinT16 [192][192] f16 (= W_in transposed)
__global__ __launch_bounds__(256) void k_projin(const _Float16* __restrict__ A,
                                                const _Float16* __restrict__ Bt,
                                                const float* __restrict__ b_in,
                                                float* __restrict__ x){
  const int gw   = (blockIdx.x * blockDim.x + threadIdx.x) >> 5; // global wave id: 0..767
  const int lane = threadIdx.x & 31;
  const int half = lane >> 4, n16 = lane & 15;
  const int rt = gw / 12, ct = gw % 12;
  const int r0 = rt * 16, c0 = ct * 16;
  v8f acc = {};
  for (int k0 = 0; k0 < C_DIM; k0 += 32){
    v16h a = load_nt16(A,  r0 + n16, C_DIM, k0, lane);
    v16h b = load_nt16(Bt, c0 + n16, C_DIM, k0, lane);
    acc = wmma_f16(a, b, acc);
  }
  const int n = c0 + n16;
  for (int i = 0; i < 8; i++){
    const int t = r0 + i + 8 * half;
    x[t * C_DIM + n] = silu_f(acc[i] + b_in[n]);
  }
}

// ---------------- Conv1d(C->8,k=3,pad=1) + LayerNorm(8) + SiLU -> xf[t][8] ----------------
__global__ void k_convln(const float* __restrict__ x, const float* __restrict__ conv_w,
                         const float* __restrict__ conv_b, const float* __restrict__ gamma,
                         const float* __restrict__ beta, float* __restrict__ xf){
  const int t = blockIdx.x * blockDim.x + threadIdx.x;
  if (t >= T_TEXT) return;
  float o[8];
  for (int f = 0; f < 8; f++){
    float s = conv_b[f];
    const float* wf = conv_w + f * C_DIM * 3;
    for (int tap = 0; tap < 3; tap++){
      const int tt = t + tap - 1;
      if (tt < 0 || tt >= T_TEXT) continue;
      const float* xr = x + tt * C_DIM;
      for (int c = 0; c < C_DIM; c++) s += xr[c] * wf[c * 3 + tap];
    }
    o[f] = s;
  }
  float mu = 0.f;
  for (int f = 0; f < 8; f++) mu += o[f];
  mu *= 0.125f;
  float var = 0.f;
  for (int f = 0; f < 8; f++){ float z = o[f] - mu; var += z * z; }
  var *= 0.125f;
  const float inv = rsqrtf(var + 1e-5f);
  for (int f = 0; f < 8; f++){
    float v = (o[f] - mu) * inv * gamma[f] + beta[f];
    xf[t * 8 + f] = silu_f(v);
  }
}

// tiny 10-dim MLPs for W_proj logits and C_proj (weights staged in LDS at sW)
__device__ __forceinline__ void mlp_eval(const float* sW, float S, float E,
                                         const float* __restrict__ xf8,
                                         float* l4, float* c2){
  float f[10]; f[0] = S; f[1] = E;
#pragma unroll
  for (int i = 0; i < 8; i++) f[2 + i] = xf8[i];
  float h[10];
#pragma unroll
  for (int j = 0; j < 10; j++){
    float s = sW[100 + j];
#pragma unroll
    for (int i = 0; i < 10; i++) s += f[i] * sW[i * 10 + j];
    h[j] = silu_f(s);
  }
#pragma unroll
  for (int q = 0; q < 4; q++){
    float s = sW[150 + q];
#pragma unroll
    for (int i = 0; i < 10; i++) s += h[i] * sW[110 + i * 4 + q];
    l4[q] = s;
  }
  float g[10];
#pragma unroll
  for (int j = 0; j < 10; j++){
    float s = sW[254 + j];
#pragma unroll
    for (int i = 0; i < 10; i++) s += f[i] * sW[154 + i * 10 + j];
    g[j] = silu_f(s);
  }
#pragma unroll
  for (int p = 0; p < 2; p++){
    float s = sW[284 + p];
#pragma unroll
    for (int i = 0; i < 10; i++) s += g[i] * sW[264 + i * 2 + p];
    c2[p] = silu_f(s);
  }
}

// ---------------- fused main kernel: softmax attention + 3 WMMA GEMMs per 16-frame tile ----------------
__global__ __launch_bounds__(256) void k_main(
    const float* __restrict__ Sd, const float* __restrict__ Ed,
    const float* __restrict__ xfG,
    const _Float16* __restrict__ Hh16,     // [192][1024] f16 (B^T for WH GEMM)
    const _Float16* __restrict__ WWHT16,   // [192][768]  f16 (W_WH transposed)
    const _Float16* __restrict__ WoutT16,  // [192][192]  f16 (W_out transposed)
    const float* __restrict__ wp1, const float* __restrict__ bp1,
    const float* __restrict__ wp2, const float* __restrict__ bp2,
    const float* __restrict__ wc1, const float* __restrict__ bc1,
    const float* __restrict__ wc2, const float* __restrict__ bc2,
    const float* __restrict__ W_WC, const float* __restrict__ b_WH,
    const float* __restrict__ b_WC, const float* __restrict__ b_out,
    float* __restrict__ out, int M)
{
  constexpr int TM = 16, TK = 128;
  __shared__ alignas(16) _Float16 sP[64 * TK];        // unnormalized softmax tile [64 rows][TK]
  __shared__ alignas(16) _Float16 sB[C_DIM * TK];     // staged H chunk [192][TK] (async DMA target)
  __shared__ alignas(16) _Float16 sWH[TM * 768];      // WH rows per frame
  __shared__ alignas(16) _Float16 sO[TM * C_DIM];     // pre-proj_out activations
  __shared__ float sMaxRed[64 * 16];
  __shared__ float sMax[64];
  __shared__ float sSum[64];
  __shared__ float sWC[TM * 8];
  __shared__ float sW[288];

  const int tid  = threadIdx.x;
  const int wid  = tid >> 5;
  const int lane = tid & 31;
  const int half = lane >> 4, n16 = lane & 15;
  const int m0 = blockIdx.x * TM;

  // stage the 286 MLP weights into LDS
  for (int i = tid; i < 286; i += 256){
    float v;
    if      (i < 100) v = wp1[i];
    else if (i < 110) v = bp1[i - 100];
    else if (i < 150) v = wp2[i - 110];
    else if (i < 154) v = bp2[i - 150];
    else if (i < 254) v = wc1[i - 154];
    else if (i < 264) v = bc1[i - 254];
    else if (i < 284) v = wc2[i - 264];
    else              v = bc2[i - 284];
    sW[i] = v;
  }
  __syncthreads();

  // ---- pass A: per-(m,q) logit max over t ----
  {
    const int mloc = tid >> 4, lane16 = tid & 15;
    const float fi = (float)(m0 + mloc + 1);
    float lmax[4] = {-3.0e38f, -3.0e38f, -3.0e38f, -3.0e38f};
    for (int t = lane16; t < T_TEXT; t += 16){
      float l4[4], c2[2];
      mlp_eval(sW, fi - Sd[t], Ed[t] - fi, xfG + t * 8, l4, c2);
      for (int q = 0; q < 4; q++) lmax[q] = fmaxf(lmax[q], l4[q]);
    }
    for (int q = 0; q < 4; q++) sMaxRed[(mloc * 4 + q) * 16 + lane16] = lmax[q];
  }
  __syncthreads();
  if (tid < 64){
    float mx = -3.0e38f;
    for (int i = 0; i < 16; i++) mx = fmaxf(mx, sMaxRed[tid * 16 + i]);
    sMax[tid] = mx; sSum[tid] = 0.f;
  }
  if (tid < TM * 8) sWC[tid] = 0.f;
  __syncthreads();

  // ---- pass B: chunked exp + WMMA accumulation of WH = P @ H^T ----
  v8f acc[6] = {};
  for (int chunk = 0; chunk < T_TEXT / TK; ++chunk){
    const int tbase = chunk * TK;

    // kick off the H-chunk staging first so the DMA overlaps the MLP VALU work
    for (int e = tid; e < C_DIM * (TK / 8); e += 256){     // 16B (8-half) transfers
      const int c = e >> 4, o = (e & 15) * 8;
      const _Float16* gsrc = Hh16 + (size_t)c * T_TEXT + tbase + o;
      _Float16* ldst = &sB[c * TK + o];
#if USE_ASYNC_LDS
      async_g2l_b128(gsrc, ldst);
#else
      *(v8h*)ldst = *(const v8h*)gsrc;
#endif
    }

    for (int idx = tid; idx < TM * TK; idx += 256){
      const int mm = idx >> 7, tt = idx & (TK - 1), t = tbase + tt;
      const float fi = (float)(m0 + mm + 1);
      float l4[4], c2[2];
      mlp_eval(sW, fi - Sd[t], Ed[t] - fi, xfG + t * 8, l4, c2);
      for (int q = 0; q < 4; q++){
        const float p = __expf(l4[q] - sMax[mm * 4 + q]);
        sP[(mm * 4 + q) * TK + tt] = (_Float16)p;
        atomicAdd(&sSum[mm * 4 + q], p);
        atomicAdd(&sWC[mm * 8 + q * 2 + 0], p * c2[0]);
        atomicAdd(&sWC[mm * 8 + q * 2 + 1], p * c2[1]);
      }
    }
    if (chunk + 1 < T_TEXT / TK)  // pull next H chunk toward the WGP
      __builtin_prefetch(Hh16 + (size_t)(tid % C_DIM) * T_TEXT + tbase + TK, 0, 1);
#if USE_ASYNC_LDS
    __builtin_amdgcn_s_wait_asynccnt(0);
#endif
    __syncthreads();

    for (int j = 0; j < 6; j++){
      const int tile = wid + j * 8;          // 48 tiles = 4 row-tiles x 12 col-tiles
      const int rt = tile / 12, ct = tile % 12;
      const int r0 = rt * 16, c0 = ct * 16;
      for (int k0 = 0; k0 < TK; k0 += 32){
        v16h a = load_nt16(sP, r0 + n16, TK, k0, lane);
        v16h b = load_nt16(sB, c0 + n16, TK, k0, lane);
        acc[j] = wmma_f16(a, b, acc[j]);
      }
    }
    __syncthreads();
  }

  // ---- normalize WH rows and stash f16 ----
  for (int j = 0; j < 6; j++){
    const int tile = wid + j * 8;
    const int rt = tile / 12, ct = tile % 12;
    for (int i = 0; i < 8; i++){
      const int r = rt * 16 + i + 8 * half;          // row = m*4 + q
      const float v = acc[j][i] / sSum[r];
      sWH[(r >> 2) * 768 + (r & 3) * C_DIM + ct * 16 + n16] = (_Float16)v;
    }
  }
  __syncthreads();
  if (tid < TM * 8) sWC[tid] /= sSum[(tid >> 3) * 4 + ((tid & 7) >> 1)];
  __syncthreads();

  // ---- Otmp = WH @ W_WH + b_WH + WC @ W_WC + b_WC ----
  for (int ct = wid; ct < 12; ct += 8){
    v8f a2 = {};
    const int c0 = ct * 16;
    for (int k0 = 0; k0 < 4 * C_DIM; k0 += 32){
      v16h a = load_nt16(sWH,    n16,      768, k0, lane);
      v16h b = load_nt16(WWHT16, c0 + n16, 768, k0, lane);
      a2 = wmma_f16(a, b, a2);
    }
    for (int i = 0; i < 8; i++){
      const int mm = i + 8 * half, n = c0 + n16;
      float v = a2[i] + b_WH[n] + b_WC[n];
      for (int j2 = 0; j2 < 8; j2++) v += sWC[mm * 8 + j2] * W_WC[j2 * C_DIM + n];
      sO[mm * C_DIM + n] = (_Float16)v;
    }
  }
  __syncthreads();

  // ---- O = Otmp @ W_out + b_out, written as [C][M] ----
  for (int ct = wid; ct < 12; ct += 8){
    v8f a3 = {};
    const int c0 = ct * 16;
    for (int k0 = 0; k0 < C_DIM; k0 += 32){
      v16h a = load_nt16(sO,      n16,      C_DIM, k0, lane);
      v16h b = load_nt16(WoutT16, c0 + n16, C_DIM, k0, lane);
      a3 = wmma_f16(a, b, a3);
    }
    for (int i = 0; i < 8; i++){
      const int mg = m0 + i + 8 * half, n = c0 + n16;
      if (mg < M) out[(size_t)n * M + mg] = a3[i] + b_out[n];
    }
  }
}

__global__ void k_zero_tail(float* __restrict__ p, int n){
  int i = blockIdx.x * blockDim.x + threadIdx.x;
  if (i < n) p[i] = 0.f;
}

extern "C" void kernel_launch(void* const* d_in, const int* in_sizes, int n_in,
                              void* d_out, int out_size, void* d_ws, size_t ws_size,
                              hipStream_t stream) {
  (void)in_sizes; (void)n_in; (void)ws_size;
  const float* H      = (const float*)d_in[0];
  const int*   d      = (const int*)  d_in[1];
  // d_in[2] = src_mask (all True for this config)
  const float* W_in   = (const float*)d_in[3];
  const float* b_in   = (const float*)d_in[4];
  const float* conv_w = (const float*)d_in[5];
  const float* conv_b = (const float*)d_in[6];
  const float* gamma  = (const float*)d_in[7];
  const float* beta   = (const float*)d_in[8];
  const float* wp1    = (const float*)d_in[9];
  const float* bp1    = (const float*)d_in[10];
  const float* wp2    = (const float*)d_in[11];
  const float* bp2    = (const float*)d_in[12];
  const float* wc1    = (const float*)d_in[13];
  const float* bc1    = (const float*)d_in[14];
  const float* wc2    = (const float*)d_in[15];
  const float* bc2    = (const float*)d_in[16];
  const float* W_WH   = (const float*)d_in[17];
  const float* b_WH   = (const float*)d_in[18];
  const float* W_WC   = (const float*)d_in[19];
  const float* b_WC   = (const float*)d_in[20];
  const float* W_out  = (const float*)d_in[21];
  const float* b_out  = (const float*)d_in[22];
  float* out = (float*)d_out;

  // recover mel length: outputs are (O [192*M], ~frame_valid [M]) concatenated
  int M = (out_size % (C_DIM + 1) == 0) ? out_size / (C_DIM + 1) : out_size / C_DIM;
  if (M <= 0) return;

  // workspace layout
  char* ws = (char*)d_ws;
  float*    Sd      = (float*)(ws + 0);
  float*    Ed      = (float*)(ws + 4096);
  float*    xf      = (float*)(ws + 8192);                 // [1024][8]
  float*    x       = (float*)(ws + 40960);                // [1024][192]
  _Float16* Ht16    = (_Float16*)(ws + 827392);            // [1024][192]
  _Float16* Hh16    = (_Float16*)(ws + 1220608);           // [192][1024]
  _Float16* WinT16  = (_Float16*)(ws + 1613824);           // [192][192]
  _Float16* WWHT16  = (_Float16*)(ws + 1687552);           // [192][768]
  _Float16* WoutT16 = (_Float16*)(ws + 1982464);           // [192][192]

  // 1. duration scan
  k_scan<<<1, T_TEXT, 0, stream>>>(d, Sd, Ed);
  // 2. f16 staging / transposes
  {
    int n = C_DIM * T_TEXT;
    k_convert_f16 <<<(n + 255) / 256, 256, 0, stream>>>(H, Hh16, n);
    k_transpose_f16<<<(n + 255) / 256, 256, 0, stream>>>(H, Ht16, C_DIM, T_TEXT);
    int nw = C_DIM * C_DIM;
    k_transpose_f16<<<(nw + 255) / 256, 256, 0, stream>>>(W_in, WinT16, C_DIM, C_DIM);
    int nh = 4 * C_DIM * C_DIM;
    k_transpose_f16<<<(nh + 255) / 256, 256, 0, stream>>>(W_WH, WWHT16, 4 * C_DIM, C_DIM);
    k_transpose_f16<<<(nw + 255) / 256, 256, 0, stream>>>(W_out, WoutT16, C_DIM, C_DIM);
  }
  // 3. proj_in GEMM (WMMA): 64 x 12 tiles = 768 waves
  k_projin<<<96, 256, 0, stream>>>(Ht16, WinT16, b_in, x);
  // 4. conv + LN + SiLU
  k_convln<<<T_TEXT / 128, 128, 0, stream>>>(x, conv_w, conv_b, gamma, beta, xf);
  // 5. fused attention + output projections
  k_main<<<(M + 15) / 16, 256, 0, stream>>>(Sd, Ed, xf, Hh16, WWHT16, WoutT16,
                                            wp1, bp1, wp2, bp2, wc1, bc1, wc2, bc2,
                                            W_WC, b_WH, b_WC, b_out, out, M);
  // 6. zero the ~frame_valid tail (all frames valid for b=1)
  int tail = out_size - C_DIM * M;
  if (tail > 0)
    k_zero_tail<<<(tail + 255) / 256, 256, 0, stream>>>(out + (size_t)C_DIM * M, tail);
}
